// TinyMoE_55748675502354
// MI455X (gfx1250) — matrix operand order repaired and checked by reading, not compiled, not run on hardware
//
#include <hip/hip_runtime.h>

// ---------------------------------------------------------------------------
// TinyMoE on gfx1250 (MI455X): token-gathered, fully fused bf16 WMMA MoE FFN.
//
//   1) prep:        out = 0, xb = bf16(x), counts = 0
//   2) convert_w:   w1/w2 fp32 -> bf16 pre-swizzled into WMMA B-fragment
//                   layout [E][N/16][K/32][lane32][16]
//   3) router:      logits -> softmax -> top2, scatter token ids + gates
//   4) moe_ffn:     per (expert, 64-token tile) block:
//                   X tile (64x1024 bf16) staged via async global->LDS DMA,
//                   loop H in 128-chunks:
//                     GEMM1 (wmma bf16, 4-deep B pipeline) + bias + relu -> LDS
//                     GEMM2 (wmma bf16, double-buffered B) accumulate Y in VGPRs
//                   epilogue: (+b2) * gate, atomicAdd into out rows
// ---------------------------------------------------------------------------

#define T_TOK 8192
#define D_DIM 1024
#define H_DIM 4096
#define E_NUM 8

#define BM    64      // tokens per block (4 WMMA M-tiles)
#define MT    (BM/16) // M-tiles
#define HC    128     // H-chunk width
#define KKS   (HC/32) // GEMM2 k-steps per chunk
#define XPAD  8       // LDS padding (halves)
#define HPAD  8

typedef __attribute__((ext_vector_type(16))) __bf16 v16bf;
typedef __attribute__((ext_vector_type(8)))  float  v8f;

union FragB { uint4 q[2]; v16bf v; };

__device__ __forceinline__ v8f wmma_bf16(v16bf a, v16bf b, v8f c) {
  return __builtin_amdgcn_wmma_f32_16x16x32_bf16(
      false, a, false, b, (short)0, c, false, false);
}

// A-fragment (16x32 bf16) from row-major LDS tile (contiguous k order: lane half h holds k = 16h + s, same as B packing).
__device__ __forceinline__ v16bf load_a_frag(const __bf16* base, int stride, int lane) {
  const int m    = lane & 15;
  const int kofs = (lane >> 4) << 4;      // 0 or 16
  const __bf16* p = base + m * stride + kofs;
  FragB u;
  u.q[0] = *(const uint4*)(p);
  u.q[1] = *(const uint4*)(p + 8);
  return u.v;
}

// B-fragment (32x16 bf16) from pre-swizzled global weights: [tile][lane][16]
__device__ __forceinline__ v16bf load_b_frag(const __bf16* wsw, size_t tile, int lane) {
  const __bf16* p = wsw + (tile * 32 + (size_t)lane) * 16;
  FragB u;
  u.q[0] = *(const uint4*)(p);
  u.q[1] = *(const uint4*)(p + 8);
  return u.v;
}

// Async global -> LDS 16B DMA (CDNA5, tracked by ASYNCcnt).
__device__ __forceinline__ void async_copy_b128(unsigned lds_off, const void* gsrc) {
  asm volatile("global_load_async_to_lds_b128 %0, %1, off"
               :: "v"(lds_off), "v"(gsrc) : "memory");
}
__device__ __forceinline__ void wait_async0() {
  asm volatile("s_wait_asynccnt 0x0" ::: "memory");
}

// ---------------------------------------------------------------------------
// 1) prep: zero out + counts, convert x -> bf16
// ---------------------------------------------------------------------------
__global__ void prep_kernel(const float* __restrict__ x, __bf16* __restrict__ xb,
                            float* __restrict__ out, int* __restrict__ counts, int n) {
  int i = blockIdx.x * blockDim.x + threadIdx.x;
  if (i < n) {
    out[i] = 0.0f;
    xb[i]  = (__bf16)x[i];
  }
  if (i < E_NUM) counts[i] = 0;
}

// ---------------------------------------------------------------------------
// 2) convert weights fp32 [E][K][N] -> bf16 WMMA-B-swizzled [E][N/16][K/32][32][16]
// ---------------------------------------------------------------------------
__global__ void convert_w_kernel(const float* __restrict__ src, __bf16* __restrict__ dst,
                                 int K, int N) {
  const int tid    = blockIdx.x * blockDim.x + threadIdx.x;
  const int lane   = tid & 31;
  const int tile   = tid >> 5;
  const int ktiles = K >> 5;
  const int ntiles = N >> 4;
  const int per_e  = ktiles * ntiles;
  const int e      = tile / per_e;
  if (e >= E_NUM) return;
  const int rem = tile - e * per_e;
  const int n_t = rem / ktiles;
  const int k_t = rem - n_t * ktiles;

  const int col   = n_t * 16 + (lane & 15);
  const int krow0 = k_t * 32 + ((lane >> 4) << 4);   // +0 or +16
  const float* s = src + ((size_t)e * K + krow0) * N + col;
  __bf16*      d = dst + ((size_t)tile * 32 + lane) * 16;
#pragma unroll
  for (int j = 0; j < 16; ++j) d[j] = (__bf16)s[(size_t)j * N];
}

// ---------------------------------------------------------------------------
// 3) router: one wave per token. logits -> softmax -> top2 -> scatter lists.
// ---------------------------------------------------------------------------
__global__ void router_kernel(const float* __restrict__ x, const float* __restrict__ wr,
                              const float* __restrict__ br,
                              int* __restrict__ list, float* __restrict__ glist,
                              int* __restrict__ counts) {
  const int tok  = (blockIdx.x * blockDim.x + threadIdx.x) >> 5;
  const int lane = threadIdx.x & 31;
  if (tok >= T_TOK) return;

  const float* xr = x + (size_t)tok * D_DIM;
  float acc[E_NUM];
#pragma unroll
  for (int e = 0; e < E_NUM; ++e) acc[e] = 0.f;

  for (int d = lane; d < D_DIM; d += 32) {
    const float xv = xr[d];
    const float* w = wr + (size_t)d * E_NUM;
#pragma unroll
    for (int e = 0; e < E_NUM; ++e) acc[e] += xv * w[e];
  }
#pragma unroll
  for (int e = 0; e < E_NUM; ++e) {
#pragma unroll
    for (int off = 16; off >= 1; off >>= 1) acc[e] += __shfl_xor(acc[e], off, 32);
  }
#pragma unroll
  for (int e = 0; e < E_NUM; ++e) acc[e] += br[e];

  float mx = acc[0];
#pragma unroll
  for (int e = 1; e < E_NUM; ++e) mx = fmaxf(mx, acc[e]);
  float p[E_NUM], sum = 0.f;
#pragma unroll
  for (int e = 0; e < E_NUM; ++e) { p[e] = __expf(acc[e] - mx); sum += p[e]; }
  const float inv = 1.f / sum;

  int   i1 = 0; float v1 = p[0];
#pragma unroll
  for (int e = 1; e < E_NUM; ++e) if (p[e] > v1) { v1 = p[e]; i1 = e; }
  int   i2 = -1; float v2 = -1.f;
#pragma unroll
  for (int e = 0; e < E_NUM; ++e) if (e != i1 && p[e] > v2) { v2 = p[e]; i2 = e; }

  if (lane == 0) {
    int pos1 = atomicAdd(&counts[i1], 1);
    list[i1 * T_TOK + pos1]  = tok;
    glist[i1 * T_TOK + pos1] = v1 * inv;
    int pos2 = atomicAdd(&counts[i2], 1);
    list[i2 * T_TOK + pos2]  = tok;
    glist[i2 * T_TOK + pos2] = v2 * inv;
  }
}

// ---------------------------------------------------------------------------
// 4) fused MoE FFN. grid = (T/BM, E), block = 256 (8 waves).
// ---------------------------------------------------------------------------
__global__ __launch_bounds__(256)
void moe_ffn_kernel(const __bf16* __restrict__ xb,
                    const __bf16* __restrict__ w1b, const float* __restrict__ b1,
                    const __bf16* __restrict__ w2b, const float* __restrict__ b2,
                    const int* __restrict__ list, const float* __restrict__ glist,
                    const int* __restrict__ counts,
                    float* __restrict__ out) {
  __shared__ __bf16 Xs[BM][D_DIM + XPAD];   // 64 x 1032 bf16 = 129 KB
  __shared__ __bf16 Hs[BM][HC + HPAD];      // 64 x 136  bf16 = 17 KB
  __shared__ int    tok_s[BM];
  __shared__ float  g_s[BM];

  const int e    = blockIdx.y;
  const int n_e  = counts[e];
  const int row0 = blockIdx.x * BM;
  if (row0 >= n_e) return;                  // block-uniform exit
  const int rcnt = min(BM, n_e - row0);

  const int tid  = threadIdx.x;
  const int lane = tid & 31;
  const int w    = tid >> 5;                // wave id 0..7

  if (tid < BM) {
    if (tid < rcnt) {
      tok_s[tid] = list[e * T_TOK + row0 + tid];
      g_s[tid]   = glist[e * T_TOK + row0 + tid];
    } else {
      tok_s[tid] = 0;                       // dummy rows read token 0, gate 0
      g_s[tid]   = 0.f;
    }
  }
  __syncthreads();

  // Stage gathered X rows into LDS via async global->LDS DMA (16B chunks).
  {
    const int cpr = D_DIM / 8;              // 128 chunks of 8 halves per row
    for (int idx = tid; idx < BM * cpr; idx += 256) {
      const int r = idx / cpr;
      const int c = idx - r * cpr;
      const void* g = (const void*)(xb + (size_t)tok_s[r] * D_DIM + c * 8);
      const unsigned lds_off = (unsigned)(uintptr_t)(const void*)&Xs[r][c * 8];
      async_copy_b128(lds_off, g);
    }
  }
  wait_async0();
  __syncthreads();

  const int  ncol = lane & 15;              // C/D column within 16-tile
  const int  mofs = (lane >> 4) << 3;       // C/D row offset 0 or 8
  const int  k1t  = D_DIM / 32;             // W1 k-tiles (32)
  const int  k2t  = H_DIM / 32;             // W2 k-tiles (128)

  // Y accumulators: this wave owns D-columns [w*128, w*128+128)
  v8f acc[MT][8];
#pragma unroll
  for (int mt = 0; mt < MT; ++mt)
#pragma unroll
    for (int nt = 0; nt < 8; ++nt)
      acc[mt][nt] = (v8f){0.f, 0.f, 0.f, 0.f, 0.f, 0.f, 0.f, 0.f};

  for (int hc = 0; hc < H_DIM; hc += HC) {
    // ---- GEMM1: this wave computes H columns [hc + w*16, +16) over 64 rows
    const int n1 = hc + w * 16;
    v8f c1[MT];
#pragma unroll
    for (int mt = 0; mt < MT; ++mt)
      c1[mt] = (v8f){0.f, 0.f, 0.f, 0.f, 0.f, 0.f, 0.f, 0.f};
    const size_t bt1 = ((size_t)e * (H_DIM / 16) + (n1 >> 4)) * k1t;

    // 4-deep rotating B-fragment pipeline.
    v16bf bb[4];
#pragma unroll
    for (int i = 0; i < 4; ++i) bb[i] = load_b_frag(w1b, bt1 + i, lane);

#pragma unroll 4
    for (int kt = 0; kt < k1t; ++kt) {
#pragma unroll
      for (int mt = 0; mt < MT; ++mt) {
        const v16bf a = load_a_frag(&Xs[mt * 16][kt * 32], D_DIM + XPAD, lane);
        c1[mt] = wmma_bf16(a, bb[kt & 3], c1[mt]);
      }
      // refill this slot with tile kt+4 (wrap at end; redundant loads harmless)
      bb[kt & 3] = load_b_frag(w1b, bt1 + ((kt + 4) & (k1t - 1)), lane);
    }

    const float bias1 = b1[(size_t)e * H_DIM + n1 + ncol];
#pragma unroll
    for (int mt = 0; mt < MT; ++mt)
#pragma unroll
      for (int r = 0; r < 8; ++r) {
        float v = c1[mt][r] + bias1;
        v = v > 0.f ? v : 0.f;              // relu
        Hs[mt * 16 + mofs + r][w * 16 + ncol] = (__bf16)v;
      }
    __syncthreads();                        // Hs chunk ready

    // ---- GEMM2 partial: Y += Hs(64 x 128) @ W2[hc:hc+128, w*128:(w+1)*128]
    // Double-buffered B: issue all 8 B-frags of step kks+1 before the
    // 32-WMMA burst of step kks so the burst hides their latency.
    v16bf b2f[2][8];
#pragma unroll
    for (int nt = 0; nt < 8; ++nt) {
      const size_t bt = ((size_t)e * (D_DIM / 16) + (w * 8 + nt)) * k2t + (hc >> 5);
      b2f[0][nt] = load_b_frag(w2b, bt, lane);
    }
#pragma unroll
    for (int kks = 0; kks < KKS; ++kks) {
      const int cur = kks & 1;
      if (kks + 1 < KKS) {
#pragma unroll
        for (int nt = 0; nt < 8; ++nt) {
          const size_t bt =
              ((size_t)e * (D_DIM / 16) + (w * 8 + nt)) * k2t + ((hc >> 5) + kks + 1);
          b2f[cur ^ 1][nt] = load_b_frag(w2b, bt, lane);
        }
      }
      v16bf a[MT];
#pragma unroll
      for (int mt = 0; mt < MT; ++mt)
        a[mt] = load_a_frag(&Hs[mt * 16][kks * 32], HC + HPAD, lane);
#pragma unroll
      for (int nt = 0; nt < 8; ++nt)
#pragma unroll
        for (int mt = 0; mt < MT; ++mt)
          acc[mt][nt] = wmma_bf16(a[mt], b2f[cur][nt], acc[mt][nt]);
    }
    __syncthreads();                        // all waves done reading Hs
  }

  // ---- epilogue: (+b2) * gate, atomic combine into out (2 contributions/token)
#pragma unroll
  for (int nt = 0; nt < 8; ++nt) {
    const int dcol = w * 128 + nt * 16 + ncol;
    const float bias2 = b2[(size_t)e * D_DIM + dcol];
#pragma unroll
    for (int mt = 0; mt < MT; ++mt)
#pragma unroll
      for (int r = 0; r < 8; ++r) {
        const int row = mt * 16 + mofs + r;
        if (row < rcnt) {
          const float v = (acc[mt][nt][r] + bias2) * g_s[row];
          atomicAdd(out + (size_t)tok_s[row] * D_DIM + dcol, v);
        }
      }
  }
}

// ---------------------------------------------------------------------------
// host launch
// ---------------------------------------------------------------------------
extern "C" void kernel_launch(void* const* d_in, const int* in_sizes, int n_in,
                              void* d_out, int out_size, void* d_ws, size_t ws_size,
                              hipStream_t stream) {
  const float* x  = (const float*)d_in[0];   // [T, D]
  const float* wr = (const float*)d_in[1];   // [D, E]
  const float* br = (const float*)d_in[2];   // [E]
  const float* w1 = (const float*)d_in[3];   // [E, D, H]
  const float* b1 = (const float*)d_in[4];   // [E, H]
  const float* w2 = (const float*)d_in[5];   // [E, H, D]
  const float* b2 = (const float*)d_in[6];   // [E, D]
  float* out = (float*)d_out;                // [T, D]

  char*  ws  = (char*)d_ws;
  size_t off = 0;
  auto alloc = [&](size_t bytes) -> void* {
    void* p = ws + off;
    off += (bytes + 255) & ~(size_t)255;
    return p;
  };
  __bf16* xb   = (__bf16*)alloc((size_t)T_TOK * D_DIM * 2);
  __bf16* w1b  = (__bf16*)alloc((size_t)E_NUM * D_DIM * H_DIM * 2);
  __bf16* w2b  = (__bf16*)alloc((size_t)E_NUM * H_DIM * D_DIM * 2);
  int*    list = (int*)   alloc((size_t)E_NUM * T_TOK * 4);
  float*  gls  = (float*) alloc((size_t)E_NUM * T_TOK * 4);
  int*    cnts = (int*)   alloc(256);

  const int TD = T_TOK * D_DIM;
  prep_kernel<<<(TD + 255) / 256, 256, 0, stream>>>(x, xb, out, cnts, TD);

  const int wthreads = E_NUM * (H_DIM / 16) * (D_DIM / 32) * 32;
  convert_w_kernel<<<wthreads / 256, 256, 0, stream>>>(w1, w1b, D_DIM, H_DIM);
  convert_w_kernel<<<wthreads / 256, 256, 0, stream>>>(w2, w2b, H_DIM, D_DIM);

  router_kernel<<<T_TOK / 8, 256, 0, stream>>>(x, wr, br, list, gls, cnts);

  dim3 grid(T_TOK / BM, E_NUM);
  moe_ffn_kernel<<<grid, 256, 0, stream>>>(xb, w1b, b1, w2b, b2, list, gls, cnts, out);
}
